// MultiHeadGatedAttention_66340064854149
// MI455X (gfx1250) — matrix operand order repaired and checked by reading, not compiled, not run on hardware
//
#include <hip/hip_runtime.h>
#include <hip/hip_bf16.h>
#include <math.h>

typedef __attribute__((ext_vector_type(16))) __bf16 v16bf;
typedef __attribute__((ext_vector_type(8)))  __bf16 v8bf;
typedef __attribute__((ext_vector_type(8)))  float  v8f;
typedef __attribute__((ext_vector_type(4)))  unsigned int v4u;
typedef __attribute__((ext_vector_type(8)))  int    v8i;
typedef __attribute__((ext_vector_type(4)))  int    v4i;

#define N_P   50000
#define EMB   1024
#define HID   256
#define NH    16
#define KSTEPS (EMB / 32)     // 32 k-steps of 32
#define DTILES (HID / 16)     // 16 column tiles of 16
#define HEAD_WSZ (DTILES * KSTEPS * 512)  // bf16 elems per head per matrix = 262144

#define MROWS  128            // patch rows per block
#define MSUB   8              // 16-row subtiles per block
#define NBLK   ((N_P + MROWS - 1) / MROWS)   // 391

// dynamic LDS: x tile (256KB bf16) + logit accumulators (8KB)
#define LDS_XT_BYTES   (MROWS * EMB * 2)
#define LDS_LACC_BYTES (8 * 2 * MROWS * 4)
#define LDS_TOTAL      (LDS_XT_BYTES + LDS_LACC_BYTES)

// ---------------- ws layout ----------------
#define OFF_LOGITS 0              // H*N = 800000 floats
#define OFF_S      800000         // N floats
#define OFF_PARTW  851968         // 128*1024 floats
#define OFF_PARTX  983040         // 128*1024 floats
#define OFF_WBF_F  1114112        // bf16 weights start here (cast)
#define OFF_XB_B   21233664       // byte offset of bf16 x copy (100MB)
#define NSLICE     128
#define RPS        391            // ceil(50000/128)

// ============================================================
// Kernel 0: convert x (f32) -> bf16 copy, row-major [N, E].
// ============================================================
__global__ __launch_bounds__(256) void convert_x_kernel(
    const float* __restrict__ x, __bf16* __restrict__ xb)
{
    size_t i = ((size_t)blockIdx.x * blockDim.x + threadIdx.x) * 8;
    float4 f0 = *reinterpret_cast<const float4*>(x + i);
    float4 f1 = *reinterpret_cast<const float4*>(x + i + 4);
    v8bf o;
    o[0] = (__bf16)f0.x; o[1] = (__bf16)f0.y;
    o[2] = (__bf16)f0.z; o[3] = (__bf16)f0.w;
    o[4] = (__bf16)f1.x; o[5] = (__bf16)f1.y;
    o[6] = (__bf16)f1.z; o[7] = (__bf16)f1.w;
    *reinterpret_cast<v8bf*>(xb + i) = o;
}

// ============================================================
// Kernel 1: pack WV/WU (f32 [H,E,Hd]) into bf16 B-fragment order.
// ============================================================
__global__ __launch_bounds__(256) void pack_weights_kernel(
    const float* __restrict__ WV, const float* __restrict__ WU,
    __bf16* __restrict__ dstV, __bf16* __restrict__ dstU)
{
    int t = blockIdx.x * blockDim.x + threadIdx.x;       // 0 .. 2*262144-1
    int mat  = t >> 18;
    int L    = t & 262143;
    int lane = L & 31;
    int ks   = (L >> 5)  & 31;
    int dt   = (L >> 10) & 15;
    int h    =  L >> 14;
    const float* src = mat ? WU : WV;
    __bf16*      dst = (mat ? dstU : dstV) + (size_t)L * 16;
    int n  = lane & 15;                 // output column within tile
    int kb = (lane < 16) ? 0 : 16;      // K half per half-wave
    size_t base = (size_t)h * EMB * HID + (size_t)(dt * 16 + n);
#pragma unroll
    for (int i = 0; i < 16; ++i) {
        int k = ks * 32 + (kb >> 1) + (i < 8 ? i : 8 + i);
        dst[i] = (__bf16)src[base + (size_t)k * HID];
    }
}

// ============================================================
// Kernel 2: gated attention logits via WMMA bf16.
// Block = 128 patches, staged into LDS by the Tensor Data Mover;
// 8 waves, wave w -> heads {w, w+8}; one d-tile per pass so each
// B fragment in registers feeds 8 WMMAs (one per M-subtile).
// ============================================================
__global__ __launch_bounds__(256) void gated_logits_kernel(
    const __bf16* __restrict__ xb,
    const __bf16* __restrict__ WVb, const __bf16* __restrict__ WUb,
    const float* __restrict__ bV, const float* __restrict__ bU,
    const float* __restrict__ wA, const float* __restrict__ bA,
    float* __restrict__ logits)
{
    extern __shared__ char smem[];
    __bf16* xt   = (__bf16*)smem;                       // [128][1024] bf16
    float*  lacc = (float*)(smem + LDS_XT_BYTES);       // [8 waves][2 heads][128 rows]

    const int t     = threadIdx.x;
    const int nbase = blockIdx.x * MROWS;

    // ---- TDM: DMA the 128x1024 bf16 tile into LDS (rows beyond N read 0) ----
    if (t < 32) {
        unsigned long long ga = (unsigned long long)(const void*)xb
                              + (unsigned long long)nbase * (EMB * 2);
        unsigned int rem = (unsigned int)(N_P - nbase);   // remaining rows
        v4u g0;
        g0[0] = 1u;                                       // count=1, user mode
        g0[1] = 0u;                                       // lds_addr = 0 (tile at LDS base)
        g0[2] = (unsigned int)(ga & 0xFFFFFFFFu);         // global_addr[31:0]
        g0[3] = (unsigned int)((ga >> 32) & 0x01FFFFFFu)  // global_addr[56:32]
              | (2u << 30);                               // type = 2 ("image")
        v8i g1;
        g1[0] = (int)(1u << 16);                          // data_size = 1 -> 2 bytes
        g1[1] = (int)((EMB & 0xFFFFu) << 16);             // tensor_dim0[15:0] = 1024
        g1[2] = (int)((rem & 0xFFFFu) << 16);             // tensor_dim1[15:0] = rem
        g1[3] = (int)(((rem >> 16) & 0xFFFFu)             // tensor_dim1[31:16]
              | ((unsigned)EMB << 16));                   // tile_dim0 = 1024
        g1[4] = (int)MROWS;                               // tile_dim1 = 128, tile_dim2 = 0
        g1[5] = (int)EMB;                                 // tensor_dim0_stride = 1024
        g1[6] = 0;
        g1[7] = 0;
        v4i gz = {0, 0, 0, 0};
#if defined(__clang_major__) && (__clang_major__ >= 23)
        v8i gz8 = {0, 0, 0, 0, 0, 0, 0, 0};
        __builtin_amdgcn_tensor_load_to_lds(g0, g1, gz, gz, gz8, 0);
#else
        __builtin_amdgcn_tensor_load_to_lds(g0, g1, gz, gz, 0);
#endif
        __builtin_amdgcn_s_wait_tensorcnt(0);
    }
    __syncthreads();

    const int wave  = t >> 5;
    const int lane  = t & 31;
    const int col   = lane & 15;            // A row / C column / B column
    const int kb    = (lane >= 16) ? 8 : 0; // A K base per half-wave
    const int rbase = (lane >= 16) ? 8 : 0; // C rows covered by this half-wave

    union AFrag { v16bf v; v8bf h2[2]; };

#pragma unroll
    for (int hh = 0; hh < 2; ++hh) {
        const int h = wave + hh * 8;
        const size_t hbase = (size_t)h * HEAD_WSZ;
        float* lrow = &lacc[(wave * 2 + hh) * MROWS];   // wave-private

        if (col == 0) {
#pragma unroll
            for (int m = 0; m < MSUB; ++m)
#pragma unroll
                for (int r = 0; r < 8; ++r)
                    lrow[m * 16 + rbase + r] = 0.f;
        }

        for (int dt = 0; dt < DTILES; ++dt) {           // one d-tile per pass
            v8f aV[MSUB] = {};
            v8f aU[MSUB] = {};
            for (int ks = 0; ks < KSTEPS; ++ks) {
                const size_t fb = hbase + (size_t)(dt * KSTEPS + ks) * 512 + lane * 16;
                v16bf bv = *reinterpret_cast<const v16bf*>(WVb + fb);
                v16bf bu = *reinterpret_cast<const v16bf*>(WUb + fb);
                const int koff = ks * 32 + kb;
#pragma unroll
                for (int m = 0; m < MSUB; ++m) {
                    AFrag af;
                    const __bf16* arow = &xt[(m * 16 + col) * EMB + koff];
                    af.h2[0] = *reinterpret_cast<const v8bf*>(arow);
                    af.h2[1] = *reinterpret_cast<const v8bf*>(arow + 16);
                    aV[m] = __builtin_amdgcn_wmma_f32_16x16x32_bf16(
                        false, af.v, false, bv, (short)0, aV[m], false, false);
                    aU[m] = __builtin_amdgcn_wmma_f32_16x16x32_bf16(
                        false, af.v, false, bu, (short)0, aU[m], false, false);
                }
            }
            // epilogue: branchless gate, project onto wA, reduce over columns
            const int d   = dt * 16 + col;
            const float bvd = bV[h * HID + d];
            const float bud = bU[h * HID + d];
            const float wad = wA[h * HID + d];
#pragma unroll
            for (int m = 0; m < MSUB; ++m) {
#pragma unroll
                for (int r = 0; r < 8; ++r) {
                    float zv = aV[m][r] + bvd;
                    float zu = aU[m][r] + bud;
                    // tanh(z) = 1 - 2/(exp(2z)+1); sigmoid(z) = 1/(1+exp(-z))
                    float vv = 1.0f - 2.0f * __builtin_amdgcn_rcpf(__expf(2.0f * zv) + 1.0f);
                    float uu = __builtin_amdgcn_rcpf(1.0f + __expf(-zu));
                    float p  = vv * uu * wad;
                    p += __shfl_xor(p, 8, 32);
                    p += __shfl_xor(p, 4, 32);
                    p += __shfl_xor(p, 2, 32);
                    p += __shfl_xor(p, 1, 32);
                    if (col == 0) lrow[m * 16 + rbase + r] += p;
                }
            }
        }
        // write logits for this head (wave-private rows; ds ops in-order)
        if (col == 0) {
            const float ba = bA[h];
#pragma unroll
            for (int m = 0; m < MSUB; ++m)
#pragma unroll
                for (int r = 0; r < 8; ++r) {
                    int n = nbase + m * 16 + rbase + r;
                    if (n < N_P)
                        logits[(size_t)h * N_P + n] = lrow[m * 16 + rbase + r] + ba;
                }
        }
    }
}

// ============================================================
// Kernel 3: per-head softmax over N; block 16 writes uniform row.
// ============================================================
__global__ __launch_bounds__(256) void softmax_kernel(
    const float* __restrict__ logits, float* __restrict__ att)
{
    const int h = blockIdx.x;
    const int t = threadIdx.x;
    if (h == NH) {
        const float u = 1.0f / (float)N_P;
        for (int n = t; n < N_P; n += 256) att[(size_t)NH * N_P + n] = u;
        return;
    }
    __shared__ float red[256];
    const float* lg = logits + (size_t)h * N_P;
    float m = -INFINITY;
    for (int n = t; n < N_P; n += 256) m = fmaxf(m, lg[n]);
    red[t] = m; __syncthreads();
    for (int s = 128; s > 0; s >>= 1) {
        if (t < s) red[t] = fmaxf(red[t], red[t + s]);
        __syncthreads();
    }
    const float mx = red[0]; __syncthreads();
    float sum = 0.f;
    for (int n = t; n < N_P; n += 256) sum += __expf(lg[n] - mx);
    red[t] = sum; __syncthreads();
    for (int s = 128; s > 0; s >>= 1) {
        if (t < s) red[t] += red[t + s];
        __syncthreads();
    }
    const float inv = 1.0f / red[0];
    float* arow = att + (size_t)h * N_P;
    for (int n = t; n < N_P; n += 256) arow[n] = __expf(lg[n] - mx) * inv;
}

// ============================================================
// Kernel 4: s[n] = sum over 16 heads of A[h,n]
// ============================================================
__global__ __launch_bounds__(256) void colsum_kernel(
    const float* __restrict__ att, float* __restrict__ s)
{
    int n = blockIdx.x * blockDim.x + threadIdx.x;
    if (n >= N_P) return;
    float acc = 0.f;
#pragma unroll
    for (int h = 0; h < NH; ++h) acc += att[(size_t)h * N_P + n];
    s[n] = acc;
}

// ============================================================
// Kernel 5: deterministic two-stage reduction over x:
// partW[slice,e] = sum_n s[n]*x[n,e],  partX[slice,e] = sum_n x[n,e]
// ============================================================
__global__ __launch_bounds__(256) void wsum_kernel(
    const float* __restrict__ x, const float* __restrict__ s,
    float* __restrict__ partW, float* __restrict__ partX)
{
    const int slice = blockIdx.x;
    const int t = threadIdx.x;
    float aW[4] = {0.f, 0.f, 0.f, 0.f};
    float aX[4] = {0.f, 0.f, 0.f, 0.f};
    const int n0 = slice * RPS;
    const int n1 = (n0 + RPS < N_P) ? n0 + RPS : N_P;
    for (int n = n0; n < n1; ++n) {
        const float* xr = x + (size_t)n * EMB;
        __builtin_prefetch(xr + EMB, 0, 0);
        const float sn = s[n];
#pragma unroll
        for (int j = 0; j < 4; ++j) {
            float v = xr[t + j * 256];
            aW[j] += sn * v;
            aX[j] += v;
        }
    }
#pragma unroll
    for (int j = 0; j < 4; ++j) {
        partW[(size_t)slice * EMB + t + j * 256] = aW[j];
        partX[(size_t)slice * EMB + t + j * 256] = aX[j];
    }
}

// ============================================================
// Kernel 6: finish — reduce partials, avg-pool MLP, combine.
// ============================================================
__global__ __launch_bounds__(256) void final_kernel(
    const float* __restrict__ partW, const float* __restrict__ partX,
    const float* __restrict__ W1, const float* __restrict__ b1,
    const float* __restrict__ W2, const float* __restrict__ b2,
    float* __restrict__ outAgg)
{
    __shared__ float avg[EMB];
    __shared__ float aggs[EMB];
    __shared__ float hid[HID];
    const int t = threadIdx.x;
#pragma unroll
    for (int j = 0; j < 4; ++j) {
        const int e = t + j * 256;
        float sw = 0.f, sx = 0.f;
        for (int k = 0; k < NSLICE; ++k) {
            sw += partW[(size_t)k * EMB + e];
            sx += partX[(size_t)k * EMB + e];
        }
        aggs[e] = sw;
        avg[e]  = sx * (1.0f / (float)N_P);
    }
    __syncthreads();
    float hacc = b1[t];
    for (int e = 0; e < EMB; ++e) hacc += avg[e] * W1[(size_t)e * HID + t];
    hid[t] = fmaxf(hacc, 0.f);
    __syncthreads();
#pragma unroll
    for (int j = 0; j < 4; ++j) {
        const int e = t + j * 256;
        float o = b2[e];
        for (int d = 0; d < HID; ++d) o += hid[d] * W2[(size_t)d * EMB + e];
        outAgg[e] = (aggs[e] + o) * (1.0f / (float)(NH + 1));
    }
}

// ============================================================
extern "C" void kernel_launch(void* const* d_in, const int* in_sizes, int n_in,
                              void* d_out, int out_size, void* d_ws, size_t ws_size,
                              hipStream_t stream) {
    const float* x  = (const float*)d_in[0];
    const float* WV = (const float*)d_in[1];
    const float* bV = (const float*)d_in[2];
    const float* WU = (const float*)d_in[3];
    const float* bU = (const float*)d_in[4];
    const float* wA = (const float*)d_in[5];
    const float* bA = (const float*)d_in[6];
    const float* W1 = (const float*)d_in[7];
    const float* b1 = (const float*)d_in[8];
    const float* W2 = (const float*)d_in[9];
    const float* b2 = (const float*)d_in[10];

    float* outf   = (float*)d_out;
    float* outAgg = outf;             // [1024]
    float* att    = outf + EMB;       // [17, 50000]

    float* wsf    = (float*)d_ws;
    float* logits = wsf + OFF_LOGITS;
    float* svec   = wsf + OFF_S;
    float* partW  = wsf + OFF_PARTW;
    float* partX  = wsf + OFF_PARTX;
    __bf16* WVb   = (__bf16*)(wsf + OFF_WBF_F);
    __bf16* WUb   = WVb + (size_t)NH * HEAD_WSZ;
    __bf16* xb    = (__bf16*)((char*)d_ws + OFF_XB_B);

    // 0) x -> bf16 copy (feeds the TDM in kernel 2)
    convert_x_kernel<<<(N_P * EMB / 8) / 256, 256, 0, stream>>>(x, xb);
    // 1) pack weights to bf16 fragment layout
    pack_weights_kernel<<<2048, 256, 0, stream>>>(WV, WU, WVb, WUb);
    // 2) gated logits via WMMA, 128-row tiles staged by TDM, 264KB dynamic LDS
    gated_logits_kernel<<<NBLK, 256, LDS_TOTAL, stream>>>(xb, WVb, WUb, bV, bU, wA, bA, logits);
    // 3) per-head softmax + uniform row
    softmax_kernel<<<NH + 1, 256, 0, stream>>>(logits, att);
    // 4) head-collapsed attention weights
    colsum_kernel<<<(N_P + 255) / 256, 256, 0, stream>>>(att, svec);
    // 5) deterministic weighted reduction over x
    wsum_kernel<<<NSLICE, 256, 0, stream>>>(x, svec, partW, partX);
    // 6) final combine + avg-pool MLP
    final_kernel<<<1, 256, 0, stream>>>(partW, partX, W1, b1, W2, b2, outAgg);
}
